// LinearAttention_23665269801325
// MI455X (gfx1250) — matrix operand order repaired and checked by reading, not compiled, not run on hardware
//
#include <hip/hip_runtime.h>
#include <hip/hip_bf16.h>
#include <cstdint>
#include <cstddef>

// ---------------------------------------------------------------------------
// Types / helpers
// ---------------------------------------------------------------------------
typedef __bf16 bf16;
typedef bf16  v16bf __attribute__((ext_vector_type(16)));
typedef float v8f   __attribute__((ext_vector_type(8)));

struct U4x2 { uint4 a, b; };   // 32B carrier for a v16bf fragment

__device__ __forceinline__ bf16 f2bf(float f) {
    unsigned u = __builtin_bit_cast(unsigned, f);
    u += 0x7FFFu + ((u >> 16) & 1u);                 // round-to-nearest-even
    return __builtin_bit_cast(bf16, (unsigned short)(u >> 16));
}
__device__ __forceinline__ unsigned pk2(float lo, float hi) {
    return (unsigned)__builtin_bit_cast(unsigned short, f2bf(lo))
         | ((unsigned)__builtin_bit_cast(unsigned short, f2bf(hi)) << 16);
}
__device__ __forceinline__ float bflo(unsigned u) {
    return __builtin_bit_cast(float, u << 16);
}
__device__ __forceinline__ float bfhi(unsigned u) {
    return __builtin_bit_cast(float, u & 0xFFFF0000u);
}

__device__ __forceinline__ v8f wmma_bf16(v16bf a, v16bf b, v8f c) {
    // D(f32 16x16) = A(bf16 16x32) * B(bf16 32x16) + C
    return __builtin_amdgcn_wmma_f32_16x16x32_bf16(
        false, a, false, b, (short)0, c, false, false);
}

// A fragment: 16x32 (MxK) bf16, row-major, lda multiple of 8 elements (16B).
// Lane L<16: row M=L, K=0..7 & 16..23;  lane L>=16: row M=L-16, K=8..15 & 24..31.
// Two contiguous 16B runs per lane -> 2x ds_load_b128.
__device__ __forceinline__ v16bf load_a_frag(const bf16* At, int lda, int lane) {
    const int m  = lane & 15;
    const int kb = (lane >> 4) * 8;
    U4x2 s;
    s.a = *reinterpret_cast<const uint4*>(At + m * lda + kb);
    s.b = *reinterpret_cast<const uint4*>(At + m * lda + 16 + kb);
    return __builtin_bit_cast(v16bf, s);
}

// B fragment from a TRANSPOSED tile Bt[n][k] (bf16 row-major, ldb mult of 8).
// Lane L<16: col N=L, K=0..7 & 16..23;  lane L>=16: col N=L-16, K=8..15 & 24..31
// (matches A fragment) -> 2x ds_load_b128.
__device__ __forceinline__ v16bf load_b_frag_t(const bf16* Bt, int ldb, int lane) {
    const int n    = lane & 15;
    const int koff = (lane >> 4) * 8;
    U4x2 s;
    s.a = *reinterpret_cast<const uint4*>(Bt + n * ldb + koff);
    s.b = *reinterpret_cast<const uint4*>(Bt + n * ldb + 16 + koff);
    return __builtin_bit_cast(v16bf, s);
}

// ---------------------------------------------------------------------------
// Kernel 1 & 4: tiled WMMA GEMM  C[M,N] = A[M,K] * B[K,N]  (+bias, +elu epi)
// 256 threads = 8 waves; block tile 128x128x32; wave tile 64x32 (4x2 frags).
// ---------------------------------------------------------------------------
template <typename AT, bool ELU_QK, bool OUT_BF16>
__global__ __launch_bounds__(256) void gemm_wmma_kernel(
    const AT* __restrict__ A, const float* __restrict__ B,
    void* __restrict__ C, const float* __restrict__ bias,
    int M, int N, int K)
{
    constexpr int BM = 128, BN = 128, BK = 32;
    constexpr int LDA = BK + 8;          // 40 bf16 = 80B rows (16B aligned)
    constexpr int LDB = BK + 8;          // Bt rows: 40 bf16
    __shared__ alignas(16) bf16 As[BM][LDA];
    __shared__ alignas(16) bf16 Bt[BN][LDB];   // transposed: Bt[n][k]

    const int t = threadIdx.x, lane = t & 31, w = t >> 5;
    const int wm = w >> 2, wn = w & 3;                 // 2 x 4 wave grid
    const int gm0 = blockIdx.y * BM, gn0 = blockIdx.x * BN;

    v8f acc[4][2] = {};

    const int arow = t >> 1, acb = (t & 1) * 16;       // A: 128 rows x 32 cols
    const int bkp  = t & 15, bcb = (t >> 4) * 8;       // B: 16 row-pairs x 8 cols

    for (int k0 = 0; k0 < K; k0 += BK) {
        { // ---- stage A tile (row-major bf16, vector stores) ----
            const AT* src = &A[(size_t)(gm0 + arow) * K + k0 + acb];
            unsigned d[8];
            if constexpr (__is_same(AT, float)) {
                const float4* s4 = reinterpret_cast<const float4*>(src);
                float4 f0 = s4[0], f1 = s4[1], f2 = s4[2], f3 = s4[3];
                d[0]=pk2(f0.x,f0.y); d[1]=pk2(f0.z,f0.w);
                d[2]=pk2(f1.x,f1.y); d[3]=pk2(f1.z,f1.w);
                d[4]=pk2(f2.x,f2.y); d[5]=pk2(f2.z,f2.w);
                d[6]=pk2(f3.x,f3.y); d[7]=pk2(f3.z,f3.w);
            } else {
                const uint4* s4 = reinterpret_cast<const uint4*>(src);
                uint4 u0 = s4[0], u1 = s4[1];
                d[0]=u0.x; d[1]=u0.y; d[2]=u0.z; d[3]=u0.w;
                d[4]=u1.x; d[5]=u1.y; d[6]=u1.z; d[7]=u1.w;
            }
            uint4* dst = reinterpret_cast<uint4*>(&As[arow][acb]);
            dst[0] = make_uint4(d[0], d[1], d[2], d[3]);
            dst[1] = make_uint4(d[4], d[5], d[6], d[7]);
        }
        { // ---- stage B tile transposed: pair-pack rows 2kp,2kp+1 ----
            const float* r0 = &B[(size_t)(k0 + 2 * bkp) * N + gn0 + bcb];
            const float* r1 = r0 + N;
            const float4* p0 = reinterpret_cast<const float4*>(r0);
            const float4* p1 = reinterpret_cast<const float4*>(r1);
            float4 a0 = p0[0], a1 = p0[1], b0 = p1[0], b1 = p1[1];
            float ra[8] = {a0.x,a0.y,a0.z,a0.w, a1.x,a1.y,a1.z,a1.w};
            float rb[8] = {b0.x,b0.y,b0.z,b0.w, b1.x,b1.y,b1.z,b1.w};
            unsigned* bt32 = reinterpret_cast<unsigned*>(&Bt[0][0]);
#pragma unroll
            for (int j = 0; j < 8; ++j)          // Bt[bcb+j][2kp..2kp+1]
                bt32[(bcb + j) * (LDB / 2) + bkp] = pk2(ra[j], rb[j]);
        }
        if (k0 + BK < K)   // hint next A tile into L2 (global_prefetch_b8)
            __builtin_prefetch(&A[(size_t)(gm0 + arow) * K + k0 + BK + acb], 0, 1);
        __syncthreads();

        v16bf af[4], bfr[2];
#pragma unroll
        for (int mt = 0; mt < 4; ++mt)
            af[mt] = load_a_frag(&As[wm * 64 + mt * 16][0], LDA, lane);
#pragma unroll
        for (int nt = 0; nt < 2; ++nt)
            bfr[nt] = load_b_frag_t(&Bt[wn * 32 + nt * 16][0], LDB, lane);
#pragma unroll
        for (int mt = 0; mt < 4; ++mt)
#pragma unroll
            for (int nt = 0; nt < 2; ++nt)
                acc[mt][nt] = wmma_bf16(af[mt], bfr[nt], acc[mt][nt]);
        __syncthreads();
    }

    // Epilogue. C-frag layout: row = r + (lane>=16)*8, col = lane&15.
    const int roff = (lane >> 4) * 8, ccol = lane & 15;
#pragma unroll
    for (int mt = 0; mt < 4; ++mt)
#pragma unroll
        for (int nt = 0; nt < 2; ++nt)
#pragma unroll
            for (int r = 0; r < 8; ++r) {
                const int row = gm0 + wm * 64 + mt * 16 + roff + r;
                const int col = gn0 + wn * 32 + nt * 16 + ccol;
                float v = acc[mt][nt][r];
                if constexpr (ELU_QK) {                 // elu(x)+1 on q,k cols
                    if (col < 2048) v = (v > 0.f) ? (v + 1.f) : __expf(v);
                }
                if (bias) v += bias[col];
                if constexpr (OUT_BF16)
                    ((bf16*)C)[(size_t)row * N + col] = f2bf(v);
                else
                    ((float*)C)[(size_t)row * N + col] = v;
            }
}

// ---------------------------------------------------------------------------
// Kernel 2: per (b,h)   kv[64,64] = k'^T v   (K = 4096),  z[64] = sum_n k'
// 128 threads = 4 waves; wave w owns kv rows d in [16w,16w+16), all 64 cols.
// k and v staged TRANSPOSED (kts[d][n], vt[e][n]) so all frags are b128 loads.
// kv written to global transposed+packed: kvT[head][e][d].
// ---------------------------------------------------------------------------
__global__ __launch_bounds__(128) void kv_z_kernel(
    const bf16* __restrict__ qkv, float* __restrict__ z_out,
    bf16* __restrict__ kvT_out)
{
    const int head = blockIdx.x;                 // b*16 + h
    const int b = head >> 4, h = head & 15;
    const bf16* kbase = qkv + (size_t)b * 4096 * 3072 + 1024 + h * 64;
    const bf16* vbase = qkv + (size_t)b * 4096 * 3072 + 2048 + h * 64;

    constexpr int LDT = 40;                      // 32 n + pad, 80B rows
    __shared__ alignas(16) bf16 kts[64][LDT];    // k^T : [d][n_local]
    __shared__ alignas(16) bf16 vt [64][LDT];    // v^T : [e][n_local]

    const int t = threadIdx.x, lane = t & 31, w = t >> 5;
    const int np = t & 15, db = (t >> 4) * 8;    // stage: n-pair x 8-d block
    const int d0 = w * 16;

    v8f   acc[4] = {};
    float zacc   = 0.f;

    for (int n0 = 0; n0 < 4096; n0 += 32) {
        // ---- stage k^T and v^T (pair-interleave rows n=2np, 2np+1) ----
        const uint4 ku0 = *reinterpret_cast<const uint4*>(
            kbase + (size_t)(n0 + 2 * np) * 3072 + db);
        const uint4 ku1 = *reinterpret_cast<const uint4*>(
            kbase + (size_t)(n0 + 2 * np + 1) * 3072 + db);
        const uint4 vu0 = *reinterpret_cast<const uint4*>(
            vbase + (size_t)(n0 + 2 * np) * 3072 + db);
        const uint4 vu1 = *reinterpret_cast<const uint4*>(
            vbase + (size_t)(n0 + 2 * np + 1) * 3072 + db);
        unsigned ka[4] = {ku0.x,ku0.y,ku0.z,ku0.w};
        unsigned kb_[4]= {ku1.x,ku1.y,ku1.z,ku1.w};
        unsigned va[4] = {vu0.x,vu0.y,vu0.z,vu0.w};
        unsigned vb[4] = {vu1.x,vu1.y,vu1.z,vu1.w};
        unsigned* k32 = reinterpret_cast<unsigned*>(&kts[0][0]);
        unsigned* v32 = reinterpret_cast<unsigned*>(&vt[0][0]);
#pragma unroll
        for (int c = 0; c < 4; ++c) {
            // source dword c of row n holds d = db+2c (lo), db+2c+1 (hi)
            k32[(db + 2*c    ) * (LDT/2) + np] = (ka[c] & 0xFFFFu)  | (kb_[c] << 16);
            k32[(db + 2*c + 1) * (LDT/2) + np] = (ka[c] >> 16) | (kb_[c] & 0xFFFF0000u);
            v32[(db + 2*c    ) * (LDT/2) + np] = (va[c] & 0xFFFFu)  | (vb[c] << 16);
            v32[(db + 2*c + 1) * (LDT/2) + np] = (va[c] >> 16) | (vb[c] & 0xFFFF0000u);
        }
        __syncthreads();

        if (t < 64) {                            // z partial: row d=t of k^T
            const uint4* kr = reinterpret_cast<const uint4*>(&kts[t][0]);
#pragma unroll
            for (int c = 0; c < 4; ++c) {
                uint4 u = kr[c];
                zacc += bflo(u.x) + bfhi(u.x) + bflo(u.y) + bfhi(u.y)
                      + bflo(u.z) + bfhi(u.z) + bflo(u.w) + bfhi(u.w);
            }
        }

        v16bf a = load_a_frag(&kts[d0][0], LDT, lane);   // A = k^T rows d0..
#pragma unroll
        for (int nt = 0; nt < 4; ++nt) {
            v16bf bb = load_b_frag_t(&vt[nt * 16][0], LDT, lane);
            acc[nt] = wmma_bf16(a, bb, acc[nt]);
        }
        __syncthreads();
    }

    // Store kv transposed+pair-packed: kvT[head][e][d], dword = (d even, d odd)
    const int roff = (lane >> 4) * 8, ccol = lane & 15;
    unsigned* kvT32 = reinterpret_cast<unsigned*>(kvT_out);
#pragma unroll
    for (int nt = 0; nt < 4; ++nt) {
        const int e = nt * 16 + ccol;
        const size_t base = ((size_t)head * 4096 + (size_t)e * 64 + d0 + roff) >> 1;
#pragma unroll
        for (int j = 0; j < 4; ++j)
            kvT32[base + j] = pk2(acc[nt][2 * j], acc[nt][2 * j + 1]);
    }
    if (t < 64) z_out[head * 64 + t] = zacc;
}

// ---------------------------------------------------------------------------
// Kernel 3: per (b,h, 64-row strip):
//   out[n,e] = (q'[n,:] @ kv)[e] / (q'[n,:].z + 1e-6), written into [B,N,C]
// 128 threads = 4 waves; wave w owns rows [16w,16w+16), M=64,N=64,K=64.
// kv arrives pre-transposed (kvT[e][d]) -> pure b128 staging + frag loads.
// ---------------------------------------------------------------------------
__global__ __launch_bounds__(128) void attn_apply_kernel(
    const bf16* __restrict__ qkv, const bf16* __restrict__ kvT,
    const float* __restrict__ z, bf16* __restrict__ out)
{
    const int head = blockIdx.y;                 // b*16 + h
    const int b = head >> 4, h = head & 15;
    const int n0 = blockIdx.x * 64;
    const bf16* qbase = qkv + (size_t)b * 4096 * 3072 + h * 64;

    constexpr int LDQ = 72;                      // 64 + pad, 144B rows
    __shared__ alignas(16) bf16  qs [64][LDQ];   // q' rows
    __shared__ alignas(16) bf16  kvt[64][LDQ];   // kv^T : [e][d]
    __shared__ float zs[64];
    __shared__ float dens[64];

    const int t = threadIdx.x, lane = t & 31, w = t >> 5;
    {
        const int lr = t >> 1, half = (t & 1) * 32;  // 2 threads per row
        const uint4* qg = reinterpret_cast<const uint4*>(
            qbase + (size_t)(n0 + lr) * 3072 + half);
        uint4* qd = reinterpret_cast<uint4*>(&qs[lr][half]);
        qd[0] = qg[0]; qd[1] = qg[1]; qd[2] = qg[2]; qd[3] = qg[3];

        const uint4* kg = reinterpret_cast<const uint4*>(
            kvT + (size_t)head * 4096 + (size_t)lr * 64 + half);
        uint4* kd = reinterpret_cast<uint4*>(&kvt[lr][half]);
        kd[0] = kg[0]; kd[1] = kg[1]; kd[2] = kg[2]; kd[3] = kg[3];
    }
    if (t < 64) zs[t] = z[head * 64 + t];
    __syncthreads();

    if (t < 64) {                                // denominator per row (b128)
        float s = 0.f;
        const uint4* qr = reinterpret_cast<const uint4*>(&qs[t][0]);
#pragma unroll
        for (int c = 0; c < 8; ++c) {
            uint4 u = qr[c];
            unsigned uu[4] = {u.x, u.y, u.z, u.w};
#pragma unroll
            for (int p = 0; p < 4; ++p) {
                const int d = c * 8 + p * 2;
                s += bflo(uu[p]) * zs[d] + bfhi(uu[p]) * zs[d + 1];
            }
        }
        dens[t] = s + 1e-6f;
    }

    const int m0 = w * 16;
    v8f acc[4] = {};
#pragma unroll
    for (int kk0 = 0; kk0 < 64; kk0 += 32) {
        v16bf a = load_a_frag(&qs[m0][kk0], LDQ, lane);
#pragma unroll
        for (int nt = 0; nt < 4; ++nt) {
            v16bf bb = load_b_frag_t(&kvt[nt * 16][kk0], LDQ, lane);
            acc[nt] = wmma_bf16(a, bb, acc[nt]);
        }
    }
    __syncthreads();                             // dens visible to all

    const int roff = (lane >> 4) * 8, ccol = lane & 15;
#pragma unroll
    for (int nt = 0; nt < 4; ++nt)
#pragma unroll
        for (int r = 0; r < 8; ++r) {
            const int row = m0 + roff + r;       // row within 64-strip
            const int col = nt * 16 + ccol;      // d-head column
            const float v = acc[nt][r] / dens[row];
            out[(size_t)(b * 4096 + n0 + row) * 1024 + h * 64 + col] = f2bf(v);
        }
}

// ---------------------------------------------------------------------------
// Host launcher
// ---------------------------------------------------------------------------
extern "C" void kernel_launch(void* const* d_in, const int* in_sizes, int n_in,
                              void* d_out, int out_size, void* d_ws, size_t ws_size,
                              hipStream_t stream) {
    (void)in_sizes; (void)n_in; (void)out_size; (void)ws_size;
    const float* x     = (const float*)d_in[0];   // [4,4096,1024]
    const float* W_qkv = (const float*)d_in[1];   // [1024,3072]
    const float* W_out = (const float*)d_in[2];   // [1024,1024]
    const float* b_out = (const float*)d_in[3];   // [1024]
    float* out = (float*)d_out;                   // [4,4096,1024]

    // Workspace layout (bf16 intermediates to halve HBM traffic):
    char* ws = (char*)d_ws;
    bf16*  qkv_ws  = (bf16*)ws;                               // 16384*3072 bf16 (96 MB)
    bf16*  attn_ws = (bf16*)(ws + (size_t)16384 * 3072 * 2);  // 16384*1024 bf16 (32 MB)
    bf16*  kvT_ws  = (bf16*)(ws + (size_t)16384 * 3072 * 2
                                + (size_t)16384 * 1024 * 2);  // 64 heads * 64*64 bf16
    float* z_ws    = (float*)((char*)kvT_ws + (size_t)64 * 64 * 64 * 2); // 64*64 f32

    // 1) qkv = x @ W_qkv, fused elu+1 on q,k columns, bf16 out
    gemm_wmma_kernel<float, true, true>
        <<<dim3(3072 / 128, 16384 / 128), 256, 0, stream>>>(
            x, W_qkv, qkv_ws, nullptr, 16384, 3072, 1024);

    // 2) kv^T = (k'^T v)^T, z = sum_n k'  (one workgroup per (b,h))
    kv_z_kernel<<<dim3(64), 128, 0, stream>>>(qkv_ws, z_ws, kvT_ws);

    // 3) attn = (q' kv) / (q'.z + eps), interleaved into [B,N,C] bf16
    attn_apply_kernel<<<dim3(4096 / 64, 64), 128, 0, stream>>>(
        qkv_ws, kvT_ws, z_ws, attn_ws);

    // 4) out = attn @ W_out + b_out, fp32 out
    gemm_wmma_kernel<bf16, false, false>
        <<<dim3(1024 / 128, 16384 / 128), 256, 0, stream>>>(
            attn_ws, W_out, out, b_out, 16384, 1024, 1024);
}
